// ResidualAttention_85581518340359
// MI455X (gfx1250) — compile-verified, matrix operand order checked by reading, not run on hardware
//
#include <hip/hip_runtime.h>

// ---------------- problem constants ----------------
constexpr int B     = 2;
constexpr int N     = 4096;
constexpr int QD    = 1024;
constexpr int CD    = 1024;
constexpr int H     = 8;
constexpr int DH    = 64;
constexpr int INNER = H * DH;   // 512
constexpr int NKV   = N + 1;    // 4097

// ---------------- vector types ----------------
typedef __attribute__((ext_vector_type(16))) __bf16          v16bf;
typedef __attribute__((ext_vector_type(8)))  float           v8f;
typedef __attribute__((ext_vector_type(4)))  unsigned short  v4us;
typedef __attribute__((ext_vector_type(8)))  unsigned short  v8us;
typedef __attribute__((ext_vector_type(16))) unsigned short  v16us;
typedef __attribute__((ext_vector_type(4)))  unsigned int    v4u;
typedef __attribute__((ext_vector_type(8)))  int             v8i;
typedef __attribute__((ext_vector_type(4)))  int             v4i;

// f32 -> bf16 through the native cast (lowers to hw cvt, pk-fusable)
__device__ inline unsigned short f2bf(float f) {
  union { __bf16 b; unsigned short u; } c;
  c.b = (__bf16)f;
  return c.u;
}

// Load a 16-element bf16 fragment as two contiguous 8-element (16B) runs.
// CDNA5 16-bit A/B layout: lanes 0-15 K={0..7,16..23}, lanes 16-31 K={8..15,24..31}.
__device__ inline v16bf load_frag(const unsigned short* p0, const unsigned short* p1) {
  v8us lo = *(const v8us*)p0;
  v8us hi = *(const v8us*)p1;
  v16us w = __builtin_shufflevector(lo, hi, 0,1,2,3,4,5,6,7,8,9,10,11,12,13,14,15);
  union { v16us u; v16bf b; } cvt; cvt.u = w;
  return cvt.b;
}

__device__ inline v8f wmma_bf16(v16bf a, v16bf b, v8f c) {
  return __builtin_amdgcn_wmma_f32_16x16x32_bf16(false, a, false, b, (short)0, c, false, false);
}

// ---------------- Tensor Data Mover: 2D strided tile -> LDS ----------------
// D# per CDNA5 ISA: group0 {flags, lds_addr, global_addr, type=2},
// group1 {data_size=2B, tensor dims (OOB -> zero fill), tile dims, dim0 stride}.
__device__ inline void tdm_load_2d_bf16(unsigned lds_off, const void* gptr,
                                        unsigned tensor_d0, unsigned tensor_d1,
                                        unsigned tile_d0, unsigned tile_d1,
                                        unsigned stride0) {
  unsigned long long ga = (unsigned long long)(size_t)gptr;
  v4u g0;
  g0[0] = 1u;                                                  // count=1, user mode
  g0[1] = lds_off;                                             // LDS byte address
  g0[2] = (unsigned)ga;                                        // global_addr[31:0]
  g0[3] = (unsigned)((ga >> 32) & 0x01FFFFFFu) | (2u << 30);   // addr[56:32] | type=2
  v8i g1;
  g1[0] = (int)(1u << 16);                                     // wg_mask=0, data_size=1 (2B)
  g1[1] = (int)((tensor_d0 & 0xFFFFu) << 16);                  // tensor_dim0[15:0]
  g1[2] = (int)(((tensor_d0 >> 16) & 0xFFFFu) | ((tensor_d1 & 0xFFFFu) << 16));
  g1[3] = (int)(((tensor_d1 >> 16) & 0xFFFFu) | ((tile_d0 & 0xFFFFu) << 16));
  g1[4] = (int)(tile_d1 & 0xFFFFu);                            // tile_dim1, tile_dim2=0
  g1[5] = (int)stride0;                                        // tensor_dim0_stride[31:0]
  g1[6] = 0;
  g1[7] = 0;
  v4i z4 = {0, 0, 0, 0};
#if __clang_major__ >= 23
  v8i z8 = {0, 0, 0, 0, 0, 0, 0, 0};
  __builtin_amdgcn_tensor_load_to_lds(g0, g1, z4, z4, z8, 0);
#else
  __builtin_amdgcn_tensor_load_to_lds(g0, g1, z4, z4, 0);
#endif
}

__device__ inline unsigned lds_off_of(const void* p) {
  return (unsigned)(size_t)p;   // low 32 bits of flat addr = LDS offset
}

// ---------------- kernel 1: pooled mean over N ----------------
__global__ void pooled_mean_kernel(const float* __restrict__ x, float* __restrict__ pooled) {
  int idx = blockIdx.x * blockDim.x + threadIdx.x;   // B*QD
  if (idx >= B * QD) return;
  int bb = idx / QD, d = idx - bb * QD;
  const float* p = x + (size_t)bb * N * QD + d;
  float s = 0.f;
  for (int n = 0; n < N; ++n) s += p[(size_t)n * QD];
  pooled[idx] = s * (1.0f / N);
}

// ---------------- kernel 2: bg row -> K row N, V^T col N ----------------
__global__ void bg_proj_kernel(const float* __restrict__ pooled,
                               const float* __restrict__ Wk_bg,
                               const float* __restrict__ Wv_bg,
                               unsigned short* __restrict__ kbuf,
                               unsigned short* __restrict__ vtb) {
  int idx = blockIdx.x * blockDim.x + threadIdx.x;   // 2 * B * INNER
  if (idx >= 2 * B * INNER) return;
  int which = idx / (B * INNER);
  int rem   = idx - which * (B * INNER);
  int bb = rem / INNER, c = rem - bb * INNER;
  const float* W  = which ? Wv_bg : Wk_bg;
  const float* pv = pooled + bb * QD;
  float s = 0.f;
  for (int k = 0; k < QD; ++k) s += pv[k] * W[(size_t)k * INNER + c];
  if (which) vtb[((size_t)(bb * INNER + c)) * NKV + N] = f2bf(s);   // V^T[b*INNER+c][N]
  else       kbuf[((size_t)(bb * NKV + N)) * INNER + c] = f2bf(s);  // K[b][N][c]
}

// ---------------- generic WMMA GEMM: Out = A[M,K] @ W[K,NC] (+bias) ----------------
// 256 threads = 8 waves; 128x64 tile; K step 32.
// OUT_MODE: 0 = f32 row-major, 1 = bf16 row-major (+extraRows bg remap), 2 = bf16 into V^T.
template<bool A_BF16, int OUT_MODE, bool BIAS>
__global__ __launch_bounds__(256) void gemm_bf16_wmma(
                               const void* __restrict__ Ap,
                               const float* __restrict__ W,
                               const float* __restrict__ bias,
                               void* __restrict__ Outp,
                               int M, int K, int NC,
                               int rowsPerBatch, int extraRows) {
  __shared__ unsigned short sA[128 * 32];   // [m][k]
  __shared__ unsigned short sB[64 * 32];    // [n][k]  (W transposed)
  const int tid  = threadIdx.x;
  const int lane = tid & 31;
  const int wave = tid >> 5;
  const int half = lane >> 4;
  const int l16  = lane & 15;
  const int kr0  = half * 8;
  const int kr1  = 16 + half * 8;
  const int row0 = blockIdx.x * 128;
  const int col0 = blockIdx.y * 64;

  const float*          Af = (const float*)Ap;
  const unsigned short* Ab = (const unsigned short*)Ap;

  v8f acc[4] = {};

  for (int kk = 0; kk < K; kk += 32) {
    if constexpr (A_BF16) {
      for (int e = tid; e < 128 * 32 / 8; e += 256) {       // v8us copies
        int r = e >> 2, c8 = e & 3;
        *(v8us*)&sA[r * 32 + c8 * 8] =
            *(const v8us*)(Ab + (size_t)(row0 + r) * K + kk + c8 * 8);
      }
    } else {
      for (int e = tid; e < 128 * 32 / 4; e += 256) {       // float4 -> packed bf16 b64
        int r = e >> 3, c4 = e & 7;
        const float4 a = *(const float4*)(Af + (size_t)(row0 + r) * K + kk + c4 * 4);
        v4us p = { f2bf(a.x), f2bf(a.y), f2bf(a.z), f2bf(a.w) };
        *(v4us*)&sA[r * 32 + c4 * 4] = p;
      }
    }
    for (int e = tid; e < 32 * 64 / 4; e += 256) {          // W tile, float4 along n
      int k = e >> 4, n4 = e & 15;
      const float4 w = *(const float4*)(W + (size_t)(kk + k) * NC + col0 + n4 * 4);
      sB[(n4 * 4 + 0) * 32 + k] = f2bf(w.x);
      sB[(n4 * 4 + 1) * 32 + k] = f2bf(w.y);
      sB[(n4 * 4 + 2) * 32 + k] = f2bf(w.z);
      sB[(n4 * 4 + 3) * 32 + k] = f2bf(w.w);
    }
    __syncthreads();

    int arow = wave * 16 + l16;
    v16bf afrag = load_frag(&sA[arow * 32 + kr0], &sA[arow * 32 + kr1]);
    v16bf bf[4];
#pragma unroll
    for (int f = 0; f < 4; ++f) {
      int brow = f * 16 + l16;
      bf[f] = load_frag(&sB[brow * 32 + kr0], &sB[brow * 32 + kr1]);
    }
#pragma unroll
    for (int f = 0; f < 4; ++f) acc[f] = wmma_bf16(afrag, bf[f], acc[f]);
    __syncthreads();
  }

#pragma unroll
  for (int f = 0; f < 4; ++f) {
    int col = col0 + f * 16 + l16;
    float bv = BIAS ? bias[col] : 0.0f;
#pragma unroll
    for (int r = 0; r < 8; ++r) {
      int grow = row0 + wave * 16 + r + half * 8;
      float val = acc[f][r] + bv;
      if constexpr (OUT_MODE == 0) {
        ((float*)Outp)[(size_t)grow * NC + col] = val;
      } else if constexpr (OUT_MODE == 1) {
        int orow = grow + (grow / rowsPerBatch) * extraRows;
        ((unsigned short*)Outp)[(size_t)orow * NC + col] = f2bf(val);
      } else {                                   // V^T[b*NC+col][n]
        int bb = grow / rowsPerBatch, n = grow - bb * rowsPerBatch;
        ((unsigned short*)Outp)[((size_t)bb * NC + col) * NKV + n] = f2bf(val);
      }
    }
  }
}

// ---------------- flash attention: 8 waves / 128 q-rows, TDM double-buffered tiles ----------------
__global__ __launch_bounds__(256, 1) void flash_attn_kernel(
                                  const unsigned short* __restrict__ qb,
                                  const unsigned short* __restrict__ kb,
                                  const unsigned short* __restrict__ vtb,
                                  const int* __restrict__ mask,
                                  unsigned short* __restrict__ ob) {
  __shared__ unsigned short kt[2][64 * 64];    // [j][d]   (TDM filled)
  __shared__ unsigned short vt[2][64 * 64];    // [d][j]   (TDM filled from V^T)
  __shared__ unsigned short pl[8][16 * 64];    // per-wave P staging

  const int tid  = threadIdx.x;
  const int lane = tid & 31;
  const int wave = tid >> 5;
  const int half = lane >> 4;
  const int l16  = lane & 15;
  const int kr0  = half * 8;
  const int kr1  = 16 + half * 8;
  const int b  = blockIdx.z;
  const int h  = blockIdx.y;
  const int q0 = blockIdx.x * 128;

  const size_t kbase = (size_t)b * NKV * INNER + h * DH;          // + j*INNER + d
  const size_t vbase = ((size_t)b * INNER + h * DH) * NKV;        // + d*NKV + j

  // preload this wave's 16x64 Q strip (register resident A-frags)
  const int qrow = q0 + wave * 16 + l16;
  const size_t qoff = ((size_t)(b * N + qrow)) * INNER + h * DH;
  v16bf qf0 = load_frag(qb + qoff + kr0,      qb + qoff + kr1);
  v16bf qf1 = load_frag(qb + qoff + 32 + kr0, qb + qoff + 32 + kr1);

  float m[8], l[8];
  v8f o[4] = {};
#pragma unroll
  for (int r = 0; r < 8; ++r) { m[r] = -3.0e38f; l[r] = 0.f; }

  const int ntiles = (NKV + 63) / 64;          // 65
  if (wave == 0) {                             // prefetch tile 0 into buffer 0
    tdm_load_2d_bf16(lds_off_of(&kt[0][0]), kb + kbase, 64, (unsigned)NKV, 64, 64, INNER);
    tdm_load_2d_bf16(lds_off_of(&vt[0][0]), vtb + vbase, (unsigned)NKV, 64, 64, 64, NKV);
  }

  int buf = 0;
  for (int t = 0; t < ntiles; ++t) {
    const int j0 = t * 64;
    if (wave == 0) {
      if (t + 1 < ntiles) {                    // prefetch next tile into other buffer
        const int j1 = j0 + 64;
        tdm_load_2d_bf16(lds_off_of(&kt[buf ^ 1][0]), kb + kbase + (size_t)j1 * INNER,
                         64, (unsigned)(NKV - j1), 64, 64, INNER);
        tdm_load_2d_bf16(lds_off_of(&vt[buf ^ 1][0]), vtb + vbase + j1,
                         (unsigned)(NKV - j1), 64, 64, 64, NKV);
        __builtin_amdgcn_s_wait_tensorcnt(2);  // current tile's 2 DMAs done
      } else {
        __builtin_amdgcn_s_wait_tensorcnt(0);
      }
    }
    __syncthreads();

    const unsigned short* ktb = kt[buf];
    const unsigned short* vtt = vt[buf];

    // S = Q @ K^T  (two bf16 WMMA k-steps per column-frag)
    v8f s[4] = {};
#pragma unroll
    for (int f = 0; f < 4; ++f) {
      int jr = f * 16 + l16;
      v16bf b0 = load_frag(&ktb[jr * 64 + kr0],      &ktb[jr * 64 + kr1]);
      v16bf b1 = load_frag(&ktb[jr * 64 + 32 + kr0], &ktb[jr * 64 + 32 + kr1]);
      s[f] = wmma_bf16(qf0, b0, s[f]);
      s[f] = wmma_bf16(qf1, b1, s[f]);
    }

    // scale + mask (column-only)
    float madd[4];
#pragma unroll
    for (int f = 0; f < 4; ++f) {
      int j = j0 + f * 16 + l16;
      bool valid = (j < NKV) && (mask[b * NKV + j] != 0);
      madd[f] = valid ? 0.0f : -3.0e38f;
    }

    // online softmax (row reductions across 4 frags x 16 lanes of the half)
    unsigned short* pw = pl[wave];
#pragma unroll
    for (int r = 0; r < 8; ++r) {
      float sv[4];
      float mx = -3.0e38f;
#pragma unroll
      for (int f = 0; f < 4; ++f) { sv[f] = s[f][r] * 0.125f + madd[f]; mx = fmaxf(mx, sv[f]); }
#pragma unroll
      for (int off = 8; off >= 1; off >>= 1) mx = fmaxf(mx, __shfl_xor(mx, off, 32));
      float mn = fmaxf(m[r], mx);
      float alpha = __expf(m[r] - mn);
      m[r] = mn;
      float rs = 0.f;
#pragma unroll
      for (int f = 0; f < 4; ++f) { float p = __expf(sv[f] - mn); sv[f] = p; rs += p; }
#pragma unroll
      for (int off = 8; off >= 1; off >>= 1) rs += __shfl_xor(rs, off, 32);
      l[r] = l[r] * alpha + rs;
#pragma unroll
      for (int f = 0; f < 4; ++f) {
        o[f][r] *= alpha;
        pw[(r + half * 8) * 64 + f * 16 + l16] = f2bf(sv[f]);   // stage P for A-frag reload
      }
    }

    // O += P @ V
#pragma unroll
    for (int f = 0; f < 4; ++f) {
#pragma unroll
      for (int kk = 0; kk < 64; kk += 32) {
        v16bf ap  = load_frag(&pw[l16 * 64 + kk + kr0], &pw[l16 * 64 + kk + kr1]);
        v16bf bv2 = load_frag(&vtt[(f * 16 + l16) * 64 + kk + kr0],
                              &vtt[(f * 16 + l16) * 64 + kk + kr1]);
        o[f] = wmma_bf16(ap, bv2, o[f]);
      }
    }
    __syncthreads();
    buf ^= 1;
  }

  // normalize + write bf16 attention output [B,N,INNER]
#pragma unroll
  for (int r = 0; r < 8; ++r) {
    float inv = 1.0f / l[r];
    int mrow = q0 + wave * 16 + r + half * 8;
#pragma unroll
    for (int f = 0; f < 4; ++f) {
      int col = h * DH + f * 16 + l16;
      ob[((size_t)(b * N + mrow)) * INNER + col] = f2bf(o[f][r] * inv);
    }
  }
}

// ---------------- host side ----------------
extern "C" void kernel_launch(void* const* d_in, const int* in_sizes, int n_in,
                              void* d_out, int out_size, void* d_ws, size_t ws_size,
                              hipStream_t stream) {
  const float* x      = (const float*)d_in[0];
  const float* ctx    = (const float*)d_in[1];
  const int*   mask   = (const int*)  d_in[2];
  const float* Wq     = (const float*)d_in[3];
  const float* Wk     = (const float*)d_in[4];
  const float* Wv     = (const float*)d_in[5];
  const float* Wk_bg  = (const float*)d_in[6];
  const float* Wv_bg  = (const float*)d_in[7];
  const float* Wout   = (const float*)d_in[8];
  const float* b_out  = (const float*)d_in[9];
  float*       out    = (float*)d_out;

  constexpr size_t POOLED_B = (size_t)B * QD * sizeof(float);
  constexpr size_t QB_B     = (size_t)B * N   * INNER * sizeof(unsigned short);
  constexpr size_t KB_B     = (size_t)B * NKV * INNER * sizeof(unsigned short);
  auto align256 = [](size_t v) { return (v + 255) & ~(size_t)255; };
  size_t o_pool = 0;
  size_t o_qb   = align256(o_pool + POOLED_B);
  size_t o_kb   = align256(o_qb   + QB_B);
  size_t o_vt   = align256(o_kb   + KB_B);
  size_t o_ao   = align256(o_vt   + KB_B);

  char* ws = (char*)d_ws;
  float*          pooled = (float*)         (ws + o_pool);
  unsigned short* qb     = (unsigned short*)(ws + o_qb);
  unsigned short* kb     = (unsigned short*)(ws + o_kb);
  unsigned short* vtb    = (unsigned short*)(ws + o_vt);
  unsigned short* ao     = (unsigned short*)(ws + o_ao);

  pooled_mean_kernel<<<dim3((B * QD + 255) / 256), 256, 0, stream>>>(x, pooled);
  bg_proj_kernel<<<dim3((2 * B * INNER + 255) / 256), 256, 0, stream>>>(
      pooled, Wk_bg, Wv_bg, kb, vtb);

  dim3 gproj(B * N / 128, INNER / 64);   // 64 x 8
  gemm_bf16_wmma<false, 1, false><<<gproj, 256, 0, stream>>>(
      x,   Wq, nullptr, qb,  B * N, QD, INNER, N, 0);
  gemm_bf16_wmma<false, 1, false><<<gproj, 256, 0, stream>>>(
      ctx, Wk, nullptr, kb,  B * N, CD, INNER, N, 1);
  gemm_bf16_wmma<false, 2, false><<<gproj, 256, 0, stream>>>(
      ctx, Wv, nullptr, vtb, B * N, CD, INNER, N, 0);

  flash_attn_kernel<<<dim3(N / 128, H, B), 256, 0, stream>>>(qb, kb, vtb, mask, ao);

  dim3 gout(B * N / 128, QD / 64);       // 64 x 16
  gemm_bf16_wmma<true, 0, true><<<gout, 256, 0, stream>>>(
      ao, Wout, b_out, out, B * N, INNER, QD, N, 0);
}
